// MultiHeadAttention_2070174237025
// MI455X (gfx1250) — matrix operand order repaired and checked by reading, not compiled, not run on hardware
//
#include <hip/hip_runtime.h>
#include <hip/hip_bf16.h>

typedef __attribute__((ext_vector_type(16))) _Float16 v16h;
typedef __attribute__((ext_vector_type(8)))  _Float16 v8h;
typedef __attribute__((ext_vector_type(8)))  float    v8f;
typedef __attribute__((ext_vector_type(4)))  float    v4f;

#define NHEADS  8
#define HD      64
#define CDIM    512
#define NSEQ    4096
#define GCH     16      // channels per group (512/32)

__device__ __forceinline__ v8f wmma16(v16h a, v16h b, v8f c) {
  return __builtin_amdgcn_wmma_f32_16x16x32_f16(
      /*neg_a=*/false, a, /*neg_b=*/false, b,
      /*c_mod=*/(short)0, c, /*reuse_a=*/false, /*reuse_b=*/false);
}

__device__ __forceinline__ v16h cat8(v8h lo, v8h hi) {
  return __builtin_shufflevector(lo, hi, 0,1,2,3,4,5,6,7,8,9,10,11,12,13,14,15);
}

// Async global -> LDS copy, 16 bytes per lane, tracked by ASYNCcnt.
// Generic pointers to __shared__ are {aperture_hi, lds_offset}; low 32 bits
// are exactly the LDS byte address the VDST operand wants (ISA 10.2 aperture).
__device__ __forceinline__ void async_ldg_b128(void* lds, const void* gptr) {
  unsigned l = (unsigned)(uintptr_t)lds;
  asm volatile("global_load_async_to_lds_b128 %0, %1, off"
               :: "v"(l), "v"(gptr) : "memory");
}
__device__ __forceinline__ void wait_async0() {
  asm volatile("s_wait_asynccnt 0x0" ::: "memory");
}

// A fragment (16x32 f16, row-major tile in LDS, leading dim ldk halfs).
// ISA layout: lane<16 -> row=lane,   K = {0..7, 16..23}
//             lane>=16 -> row=lane-16, K = {8..15, 24..31}
__device__ __forceinline__ v16h load_a(const _Float16* p, int ldk) {
  int lane = threadIdx.x & 31;
  int row  = lane & 15;
  int kb   = (lane >> 4) * 8;
  v8h lo = *(const v8h*)(p + row * ldk + kb);
  v8h hi = *(const v8h*)(p + row * ldk + kb + 16);
  return cat8(lo, hi);
}

// B fragment (32x16 f16) from an LDS tile stored as [n][k] (k contiguous).
// ISA layout: lane<16 -> col=lane,    K = {0..7, 16..23}
//             lane>=16 -> col=lane-16, K = {8..15, 24..31}
__device__ __forceinline__ v16h load_b(const _Float16* p, int ldk) {
  int lane = threadIdx.x & 31;
  int n    = lane & 15;
  int kb   = (lane >> 4) * 8;
  return cat8(*(const v8h*)(p + n * ldk + kb), *(const v8h*)(p + n * ldk + kb + 16));
}

// ---------------------------------------------------------------------------
// Kernel 1: GroupNorm over [B,512,64,64] -> f16 activations in [B,N,C]
// ---------------------------------------------------------------------------
__global__ __launch_bounds__(256)
void gn_kernel(const float* __restrict__ q, const float* __restrict__ gw,
               const float* __restrict__ gb, _Float16* __restrict__ x16) {
  int blk   = blockIdx.x;         // 0..63
  int batch = blk >> 5;
  int g     = blk & 31;
  const float* base = q + ((size_t)batch * CDIM + (size_t)g * GCH) * NSEQ;
  const v4f* base4 = (const v4f*)base;
  int tid = threadIdx.x;

  float s = 0.f, ss = 0.f;
  const int NV4 = GCH * NSEQ / 4;   // 16384
  for (int i = tid; i < NV4; i += 256) {
    v4f v = base4[i];
    s  += v.x + v.y + v.z + v.w;
    ss += v.x*v.x + v.y*v.y + v.z*v.z + v.w*v.w;
  }
  __shared__ float red[2][256];
  red[0][tid] = s; red[1][tid] = ss;
  __syncthreads();
  for (int st = 128; st > 0; st >>= 1) {
    if (tid < st) { red[0][tid] += red[0][tid+st]; red[1][tid] += red[1][tid+st]; }
    __syncthreads();
  }
  const float inv_n = 1.0f / (float)(GCH * NSEQ);
  float mean = red[0][0] * inv_n;
  float var  = red[1][0] * inv_n - mean * mean;
  float rinv = rsqrtf(var + 1e-5f);

  for (int i = tid; i < NV4; i += 256) {
    int lin = i * 4;
    int ch  = lin >> 12;            // /4096 (constant within float4)
    int n   = lin & 4095;
    int c   = g * GCH + ch;
    float sc = gw[c] * rinv;
    float sh = gb[c] - mean * sc;
    v4f v = base4[i];
    _Float16* o = x16 + ((size_t)batch * NSEQ + n) * CDIM + c;
    o[0*CDIM] = (_Float16)(v.x * sc + sh);
    o[1*CDIM] = (_Float16)(v.y * sc + sh);
    o[2*CDIM] = (_Float16)(v.z * sc + sh);
    o[3*CDIM] = (_Float16)(v.w * sc + sh);
  }
}

// ---------------------------------------------------------------------------
// Kernel 2: fused QKV projection. y = x @ W^T + b  (blockIdx.z picks q/k/v)
// Block tile 128x64, 8 waves (4x2), wave tile 32x32, K-step 32.
// Double-buffered LDS; A-tile staged via global_load_async_to_lds_b128 with
// the async latency hidden behind the current buffer's WMMAs.
// ---------------------------------------------------------------------------
__global__ __launch_bounds__(256)
void qkv_gemm(const _Float16* __restrict__ x16,
              const float* __restrict__ wq, const float* __restrict__ bq,
              const float* __restrict__ wk, const float* __restrict__ bk,
              const float* __restrict__ wv, const float* __restrict__ bv,
              _Float16* __restrict__ qkv) {
  int t = blockIdx.z;
  const float* W    = (t == 0) ? wq : (t == 1) ? wk : wv;
  const float* bias = (t == 0) ? bq : (t == 1) ? bk : bv;
  _Float16* out = qkv + (size_t)t * 2 * NSEQ * CDIM;
  const float oscale = (t == 0) ? 0.125f : 1.0f;   // 1/sqrt(64) folded into q

  __shared__ __align__(128) _Float16 As[2][128 * 32];
  __shared__ __align__(128) _Float16 Bs[2][64 * 32];

  int tid = threadIdx.x, wid = tid >> 5, lane = tid & 31;
  int wr = wid >> 1, wc = wid & 1;
  int row0 = blockIdx.x * 128;
  int col0 = blockIdx.y * 64;

  auto stageA = [&](int buf, int k0) {           // 128x32 halfs, async path
    int r = tid >> 1, p = (tid & 1) * 16;
    const _Float16* g = x16 + (size_t)(row0 + r) * CDIM + k0 + p;
    _Float16*       l = As[buf] + r * 32 + p;
    async_ldg_b128(l,     g);
    async_ldg_b128(l + 8, g + 8);
  };
  auto stageB = [&](int buf, int k0) {           // 64x32 f32->f16, [n][k]
    int r = tid >> 2, p = (tid & 3) * 8;
    v8f v = *(const v8f*)(W + (size_t)(col0 + r) * CDIM + k0 + p);
    v8h h;
    #pragma unroll
    for (int e = 0; e < 8; e++) h[e] = (_Float16)v[e];
    *(v8h*)(Bs[buf] + r * 32 + p) = h;
  };

  v8f acc[2][2] = {};

  stageA(0, 0);
  stageB(0, 0);
  wait_async0();
  __syncthreads();

  #pragma unroll 2
  for (int it = 0; it < CDIM / 32; it++) {
    int cur = it & 1, nxt = cur ^ 1;
    if (it + 1 < CDIM / 32) {                    // prefetch next K-step
      stageA(nxt, (it + 1) * 32);
      stageB(nxt, (it + 1) * 32);
    }
    v16h a0 = load_a(As[cur] + (wr * 32 + 0 ) * 32, 32);
    v16h a1 = load_a(As[cur] + (wr * 32 + 16) * 32, 32);
    v16h b0 = load_b(Bs[cur] + (wc * 32 + 0 ) * 32, 32);
    v16h b1 = load_b(Bs[cur] + (wc * 32 + 16) * 32, 32);
    acc[0][0] = wmma16(a0, b0, acc[0][0]);
    acc[0][1] = wmma16(a0, b1, acc[0][1]);
    acc[1][0] = wmma16(a1, b0, acc[1][0]);
    acc[1][1] = wmma16(a1, b1, acc[1][1]);
    wait_async0();                                // nxt staging complete
    __syncthreads();                              // + everyone done with cur
  }

  int half = lane >> 4, ln = lane & 15;
  #pragma unroll
  for (int mt = 0; mt < 2; mt++)
    #pragma unroll
    for (int nt = 0; nt < 2; nt++) {
      int cout = col0 + wc * 32 + nt * 16 + ln;
      float bv_ = bias[cout];
      int head = cout >> 6, d = cout & 63;
      #pragma unroll
      for (int j = 0; j < 8; j++) {
        int m  = row0 + wr * 32 + mt * 16 + j + half * 8;
        int bb = m >> 12, n = m & 4095;
        float v = (acc[mt][nt][j] + bv_) * oscale;
        out[(((size_t)bb * NHEADS + head) * NSEQ + n) * HD + d] = (_Float16)v;
      }
    }
}

// ---------------------------------------------------------------------------
// Kernel 3: flash attention. grid = (N/64, heads, B), 128 threads (4 waves).
// Double-buffered K/V tiles: async K staging + V transpose for the NEXT block
// are issued before the current block's WMMAs, one barrier per block.
// ---------------------------------------------------------------------------
__global__ __launch_bounds__(128)
void flash_attn(const _Float16* __restrict__ qkv, _Float16* __restrict__ ao) {
  const _Float16* Q = qkv;
  const _Float16* K = qkv + (size_t)2 * NSEQ * CDIM;
  const _Float16* V = qkv + (size_t)4 * NSEQ * CDIM;
  int b = blockIdx.z, h = blockIdx.y;
  size_t hb = ((size_t)b * NHEADS + h) * NSEQ * HD;
  int qr0 = blockIdx.x * 64;
  int tid = threadIdx.x, wid = tid >> 5, lane = tid & 31;
  int half = lane >> 4, ln = lane & 15;

  __shared__ __align__(128) _Float16 Kt[2][64 * 64];    // [j][d]
  __shared__ __align__(128) _Float16 Vt[2][64 * 64];    // transposed: [d][j]
  __shared__ __align__(128) _Float16 Pt[4][16 * 64];    // per-wave P tile

  auto stageK = [&](int buf, const _Float16* Kg) {      // async path
    #pragma unroll
    for (int i = 0; i < 4; i++) {
      int c = i * 128 + tid;
      int r = c >> 3, c8 = (c & 7) * 8;
      async_ldg_b128(Kt[buf] + r * 64 + c8, Kg + r * 64 + c8);
    }
  };
  auto stageV = [&](int buf, const _Float16* Vg) {      // transpose via VGPRs
    #pragma unroll
    for (int i = 0; i < 4; i++) {
      int c = i * 128 + tid;
      int r = c >> 3, c8 = (c & 7) * 8;
      v8h vv = *(const v8h*)(Vg + r * 64 + c8);
      #pragma unroll
      for (int e = 0; e < 8; e++) Vt[buf][(c8 + e) * 64 + r] = vv[e];
    }
  };

  // Q A-fragments straight from global (rows are 128B-contiguous)
  const _Float16* qrow = Q + hb + (size_t)(qr0 + wid * 16 + ln) * HD;
  int kb8 = half * 8;
  v16h qa0 = cat8(*(const v8h*)(qrow + kb8),      *(const v8h*)(qrow + kb8 + 16));
  v16h qa1 = cat8(*(const v8h*)(qrow + 32 + kb8), *(const v8h*)(qrow + 32 + kb8 + 16));

  v8f accv[4] = {};
  float mrow[8], lrow[8];
  #pragma unroll
  for (int j = 0; j < 8; j++) { mrow[j] = -3.0e38f; lrow[j] = 0.f; }

  stageK(0, K + hb);
  stageV(0, V + hb);
  wait_async0();
  __syncthreads();

  #pragma unroll 2
  for (int kt = 0; kt < NSEQ / 64; kt++) {
    int cur = kt & 1, nxt = cur ^ 1;
    if (kt + 1 < NSEQ / 64) {                   // stage next K/V block
      const _Float16* Kg = K + hb + (size_t)(kt + 1) * 64 * HD;
      const _Float16* Vg = V + hb + (size_t)(kt + 1) * 64 * HD;
      __builtin_prefetch(Kg + 64 * HD + tid * 32, 0, 0);   // block kt+2
      __builtin_prefetch(Vg + 64 * HD + tid * 32, 0, 0);
      stageK(nxt, Kg);
      stageV(nxt, Vg);
    }

    // scores S = q @ k^T (q pre-scaled): 4 col-tiles of 16 keys
    const _Float16* Kc = Kt[cur];
    v8f s[4];
    #pragma unroll
    for (int ct = 0; ct < 4; ct++) {
      v8f z = {};
      z = wmma16(qa0, load_b(Kc + ct * 16 * 64, 64), z);        // d=0..31
      z = wmma16(qa1, load_b(Kc + ct * 16 * 64 + 32, 64), z);   // d=32..63
      s[ct] = z;
    }

    // online softmax (row r = j + 8*half lives across the 16-lane half)
    float pex[4][8];
    #pragma unroll
    for (int j = 0; j < 8; j++) {
      float mx = fmaxf(fmaxf(s[0][j], s[1][j]), fmaxf(s[2][j], s[3][j]));
      #pragma unroll
      for (int msk = 8; msk >= 1; msk >>= 1) mx = fmaxf(mx, __shfl_xor(mx, msk, 32));
      float mnew = fmaxf(mrow[j], mx);
      float corr = __expf(mrow[j] - mnew);
      mrow[j] = mnew;
      float rs = 0.f;
      #pragma unroll
      for (int ct = 0; ct < 4; ct++) {
        float p = __expf(s[ct][j] - mnew);
        pex[ct][j] = p;
        rs += p;
      }
      #pragma unroll
      for (int msk = 8; msk >= 1; msk >>= 1) rs += __shfl_xor(rs, msk, 32);
      lrow[j] = lrow[j] * corr + rs;
      #pragma unroll
      for (int dt = 0; dt < 4; dt++) accv[dt][j] *= corr;
    }

    // C-layout f32 -> A-layout f16 via per-wave LDS tile
    _Float16* P = Pt[wid];
    #pragma unroll
    for (int ct = 0; ct < 4; ct++)
      #pragma unroll
      for (int j = 0; j < 8; j++)
        P[(j + half * 8) * 64 + ct * 16 + ln] = (_Float16)pex[ct][j];
    asm volatile("s_wait_dscnt 0" ::: "memory");   // CDNA5 split DS counter

    // O += P @ V  (contraction over 64 keys, 4 output d-tiles)
    const _Float16* Vc = Vt[cur];
    v16h pa0 = load_a(P, 64);
    v16h pa1 = load_a(P + 32, 64);
    #pragma unroll
    for (int dt = 0; dt < 4; dt++) {
      accv[dt] = wmma16(pa0, load_b(Vc + dt * 16 * 64, 64), accv[dt]);
      accv[dt] = wmma16(pa1, load_b(Vc + dt * 16 * 64 + 32, 64), accv[dt]);
    }
    wait_async0();                                 // next K tile landed
    __syncthreads();                               // everyone done with cur
  }

  // normalize and write head-concatenated [B,N,C] f16
  #pragma unroll
  for (int dt = 0; dt < 4; dt++)
    #pragma unroll
    for (int j = 0; j < 8; j++) {
      int n = qr0 + wid * 16 + j + half * 8;
      int c = h * 64 + dt * 16 + ln;
      float o = accv[dt][j] / lrow[j];
      ao[((size_t)b * NSEQ + n) * CDIM + c] = (_Float16)o;
    }
}

// ---------------------------------------------------------------------------
// Kernel 4: output projection + bias + residual, fp32 out in [B,C,H,W]
// ---------------------------------------------------------------------------
__global__ __launch_bounds__(256)
void out_proj(const _Float16* __restrict__ ao, const float* __restrict__ wo,
              const float* __restrict__ bo, const float* __restrict__ resid,
              float* __restrict__ out) {
  __shared__ __align__(128) _Float16 As[2][128 * 32];
  __shared__ __align__(128) _Float16 Bs[2][64 * 32];
  int tid = threadIdx.x, wid = tid >> 5, lane = tid & 31;
  int wr = wid >> 1, wc = wid & 1;
  int row0 = blockIdx.x * 128, col0 = blockIdx.y * 64;

  auto stageA = [&](int buf, int k0) {
    int r = tid >> 1, p = (tid & 1) * 16;
    const _Float16* g = ao + (size_t)(row0 + r) * CDIM + k0 + p;
    _Float16*       l = As[buf] + r * 32 + p;
    async_ldg_b128(l,     g);
    async_ldg_b128(l + 8, g + 8);
  };
  auto stageB = [&](int buf, int k0) {
    int r = tid >> 2, p = (tid & 3) * 8;
    v8f v = *(const v8f*)(wo + (size_t)(col0 + r) * CDIM + k0 + p);
    v8h hh;
    #pragma unroll
    for (int e = 0; e < 8; e++) hh[e] = (_Float16)v[e];
    *(v8h*)(Bs[buf] + r * 32 + p) = hh;
  };

  v8f acc[2][2] = {};

  stageA(0, 0);
  stageB(0, 0);
  wait_async0();
  __syncthreads();

  #pragma unroll 2
  for (int it = 0; it < CDIM / 32; it++) {
    int cur = it & 1, nxt = cur ^ 1;
    if (it + 1 < CDIM / 32) {
      stageA(nxt, (it + 1) * 32);
      stageB(nxt, (it + 1) * 32);
    }
    v16h a0 = load_a(As[cur] + (wr * 32 + 0 ) * 32, 32);
    v16h a1 = load_a(As[cur] + (wr * 32 + 16) * 32, 32);
    v16h b0 = load_b(Bs[cur] + (wc * 32 + 0 ) * 32, 32);
    v16h b1 = load_b(Bs[cur] + (wc * 32 + 16) * 32, 32);
    acc[0][0] = wmma16(a0, b0, acc[0][0]);
    acc[0][1] = wmma16(a0, b1, acc[0][1]);
    acc[1][0] = wmma16(a1, b0, acc[1][0]);
    acc[1][1] = wmma16(a1, b1, acc[1][1]);
    wait_async0();
    __syncthreads();
  }

  int half = lane >> 4, ln = lane & 15;
  #pragma unroll
  for (int mt = 0; mt < 2; mt++)
    #pragma unroll
    for (int nt = 0; nt < 2; nt++) {
      int cout = col0 + wc * 32 + nt * 16 + ln;
      float bb = bo[cout];
      #pragma unroll
      for (int j = 0; j < 8; j++) {
        int m   = row0 + wr * 32 + mt * 16 + j + half * 8;
        int bat = m >> 12, n = m & 4095;
        size_t oi = ((size_t)bat * CDIM + cout) * NSEQ + n;
        out[oi] = acc[mt][nt][j] + bb + resid[oi];
      }
    }
}

// ---------------------------------------------------------------------------
extern "C" void kernel_launch(void* const* d_in, const int* in_sizes, int n_in,
                              void* d_out, int out_size, void* d_ws, size_t ws_size,
                              hipStream_t stream) {
  const float* q   = (const float*)d_in[0];
  const float* gnw = (const float*)d_in[1];
  const float* gnb = (const float*)d_in[2];
  const float* wq  = (const float*)d_in[3];
  const float* bq  = (const float*)d_in[4];
  const float* wk  = (const float*)d_in[5];
  const float* bk  = (const float*)d_in[6];
  const float* wv  = (const float*)d_in[7];
  const float* bv  = (const float*)d_in[8];
  const float* wo  = (const float*)d_in[9];
  const float* bo  = (const float*)d_in[10];
  float* out = (float*)d_out;

  const size_t SZ = (size_t)2 * NSEQ * CDIM;   // elements of one [B,N,C] f16 tensor
  _Float16* x16 = (_Float16*)d_ws;             // 8 MB
  _Float16* qkv = x16 + SZ;                    // 24 MB (q,k,v)
  _Float16* ao  = qkv + 3 * SZ;                // 8 MB

  gn_kernel<<<dim3(64), dim3(256), 0, stream>>>(q, gnw, gnb, x16);
  qkv_gemm<<<dim3(64, 8, 3), dim3(256), 0, stream>>>(x16, wq, bq, wk, bk, wv, bv, qkv);
  flash_attn<<<dim3(64, 8, 2), dim3(128), 0, stream>>>(qkv, ao);
  out_proj<<<dim3(64, 8), dim3(256), 0, stream>>>(ao, wo, bo, q, out);
}
